// ResidualContextLSTM_64098091925584
// MI455X (gfx1250) — matrix operand order repaired and checked by reading, not compiled, not run on hardware
//
#include <hip/hip_runtime.h>
#include <stdint.h>
#include <stddef.h>

// ---------------- problem constants ----------------
#define TT 1024
#define DD 1024
#define BB 16
#define GG 4096

// ---------------- recurrent-kernel tiling ----------------
#define NWG 32                    // persistent workgroups (one WGP each)
#define HCOLS 32                  // h columns owned per WG
#define KCHUNK 64                 // K elements per TDM chunk
#define NCHUNK (DD / KCHUNK)      // 16 chunks per step
#define ROWB 144                  // LDS bytes per (U-row, chunk): 64*2 data + 16 pad
#define UBUF_BYTES (128 * ROWB)   // 18432 B per chunk buffer (128 U-rows)
#define ZBUF_OFF (2 * UBUF_BYTES) // 36864
#define SMEM_BYTES (ZBUF_OFF + 16 * 128 * 4)  // + z tile [16][128] f32 = 45056 B

typedef __bf16 bf16_t;
typedef bf16_t v16bf __attribute__((ext_vector_type(16)));
typedef float v8f __attribute__((ext_vector_type(8)));
typedef unsigned int u32x4v __attribute__((ext_vector_type(4)));
typedef int i32x4v __attribute__((ext_vector_type(4)));
typedef int i32x8v __attribute__((ext_vector_type(8)));

union FragU {
  uint4 q[2];
  v16bf v;
};

__device__ __forceinline__ unsigned short f2bf(float f) {
  union { float f; unsigned u; } c; c.f = f;
  unsigned r = 0x7fffu + ((c.u >> 16) & 1u);  // round-to-nearest-even
  return (unsigned short)((c.u + r) >> 16);
}

__device__ __forceinline__ float sigmoidf_(float v) {
  return 1.0f / (1.0f + __expf(-v));
}

__device__ __forceinline__ v8f wmma_bf16(v16bf a, v16bf b, v8f c) {
  return __builtin_amdgcn_wmma_f32_16x16x32_bf16(false, a, false, b, (short)0, c,
                                                 false, false);
}

// A fragment (16x32 bf16, row-major source). ISA 7.12.2: lanes 0-15 hold
// K = k+0..7 (VGPR0-3) and k+16..23 (VGPR4-7); lanes 16-31 are +8.
__device__ __forceinline__ v16bf load_a_frag(const unsigned short* rowk, int lane) {
  const int ko = (lane & 16) ? 8 : 0;
  FragU u;
  u.q[0] = *(const uint4*)(rowk + ko);
  u.q[1] = *(const uint4*)(rowk + ko + 16);
  return u.v;
}

// B fragment (32x16 bf16) from per-column contiguous-K storage:
// lane n (col n) lanes 0-15 -> K k+0..7, k+16..23; lanes 16-31 are +8.
__device__ __forceinline__ v16bf load_b_frag(const unsigned short* colk, int lane) {
  const int ko = (lane & 16) ? 8 : 0;
  FragU u;
  u.q[0] = *(const uint4*)(colk + ko);
  u.q[1] = *(const uint4*)(colk + ko + 16);
  return u.v;
}

// ---------------- Tensor Data Mover: one 3D tile (K x 32rows x 4gates) ----------------
// Tensor = U^T viewed as [gate(stride 1M elem)][row(stride 1K elem)][K].
// Tile = KCHUNK x 32 x 4 -> 128 LDS rows of 128B, padded +16B each (pad_interval=32DW,
// pad_amount=4DW) => row stride 144B.
__device__ __forceinline__ void tdm_issue_u(uint64_t gaddr, unsigned lds_off) {
  u32x4v g0;
  g0[0] = 1u;                                          // count=1 (valid descriptor)
  g0[1] = lds_off;                                     // LDS byte address
  g0[2] = (unsigned)(gaddr & 0xffffffffull);           // global_addr[31:0]
  g0[3] = (unsigned)((gaddr >> 32) & 0x1ffffffull) | (2u << 30);  // addr[56:32]|type=2
  i32x8v g1;
  g1[0] = (int)((1u << 16) |        // data_size = 2B
                (1u << 20) |        // pad_enable
                (4u << 22) |        // pad_interval = 32 DW (128 B)
                (3u << 25));        // pad_amount = 4 DW (16 B)
  g1[1] = (int)(1024u << 16);       // tensor_dim0[15:0] = 1024 (K extent)
  g1[2] = (int)(1024u << 16);       // tensor_dim1[15:0] = 1024 (rows per gate)
  g1[3] = (int)((unsigned)KCHUNK << 16);  // tile_dim0 = 64
  g1[4] = (int)(32u | (4u << 16));  // tile_dim1 = 32 rows, tile_dim2 = 4 gates
  g1[5] = 1024;                     // tensor_dim0_stride = 1024 elems
  g1[6] = 0;                        // dim0_stride hi | dim1_stride lo16 (1M -> 0)
  g1[7] = 16;                       // tensor_dim1_stride[47:16] = 1M >> 16
  i32x4v g2;
  g2[0] = 4;                        // tensor_dim2 = 4 gates
  g2[1] = 0;                        // tensor_dim3 unused
  g2[2] = 0x00100000;               // tensor_dim2_stride = 1M elems
  g2[3] = 0;                        // tile_dim3 = 0
  i32x4v g3;
  g3[0] = 0; g3[1] = 0; g3[2] = 0; g3[3] = 0;
  i32x8v g4;                        // trailing group (6-arg builtin form): no
  g4[0] = 0; g4[1] = 0; g4[2] = 0;  // gather/iterate/extended state in use
  g4[3] = 0; g4[4] = 0; g4[5] = 0;
  g4[6] = 0; g4[7] = 0;
  __builtin_amdgcn_tensor_load_to_lds(g0, g1, g2, g3, g4, 0);
}

// ---------------- conversion / init kernels ----------------
// x[b][t][d] f32 -> xb[(t*16+b)][d] bf16  (GEMM-friendly row order = xW step order)
__global__ void __launch_bounds__(256) k_cvt_x(const float* __restrict__ x,
                                               unsigned short* __restrict__ xb) {
  size_t i = ((size_t)blockIdx.x * 256 + threadIdx.x) * 4;
#pragma unroll
  for (int j = 0; j < 4; ++j) {
    size_t k = i + j;
    int d = (int)(k & 1023);
    int b = (int)((k >> 10) & 15);
    size_t t = k >> 14;
    xb[k] = f2bf(x[((size_t)b * TT + t) * DD + d]);
  }
}

// M[d][g] f32 -> MT[g][d] bf16 (column-major staging for B fragments)
__global__ void __launch_bounds__(256) k_cvt_t(const float* __restrict__ m,
                                               unsigned short* __restrict__ mt) {
  size_t i = ((size_t)blockIdx.x * 256 + threadIdx.x) * 4;
#pragma unroll
  for (int j = 0; j < 4; ++j) {
    size_t k = i + j;
    int d = (int)(k & 1023);
    size_t g = k >> 10;
    mt[k] = f2bf(m[(size_t)d * GG + g]);
  }
}

__global__ void __launch_bounds__(256) k_init(unsigned short* __restrict__ hbf,
                                              unsigned* __restrict__ bar) {
  int i = blockIdx.x * 256 + threadIdx.x;
  if (i < 2 * BB * DD) hbf[i] = 0;
  if (i < 2) bar[i] = 0;
}

// ---------------- phase 1: xW = xb @ W (+bias), bf16 WMMA, f32 out ----------------
// WG = 8 waves stacked in M (128 rows), each wave computes 16x64 (4 N-tiles).
// Copy-free two-set ping-pong: set A covers K-slab kb, set B covers kb+32.
// A set's registers are reloaded (not copied) right after its 4 WMMAs consume
// them, while the other set's WMMAs execute -> load latency hidden, no
// v_dual_mov rotation, no extra WMMA WAR hazard NOPs.
__global__ void __launch_bounds__(256) k_xw_gemm(const unsigned short* __restrict__ xb,
                                                 const unsigned short* __restrict__ wt,
                                                 const float* __restrict__ bias,
                                                 float* __restrict__ xw) {
  const int tid = threadIdx.x;
  const int lane = tid & 31;
  const int ln16 = lane & 15;
  const int wid = tid >> 5;
  const int m0 = (int)(blockIdx.x >> 6) * 128 + wid * 16;
  const int n0 = (int)(blockIdx.x & 63) * 64;

  v8f acc[4] = {};
  const unsigned short* arow = xb + (size_t)(m0 + ln16) * DD;
  const unsigned short* brow[4];
#pragma unroll
  for (int j = 0; j < 4; ++j)
    brow[j] = wt + (size_t)(n0 + j * 16 + ln16) * DD;

  // prologue: set A = K slab 0, set B = K slab 32
  v16bf aA = load_a_frag(arow, lane);
  v16bf bA0 = load_b_frag(brow[0], lane);
  v16bf bA1 = load_b_frag(brow[1], lane);
  v16bf bA2 = load_b_frag(brow[2], lane);
  v16bf bA3 = load_b_frag(brow[3], lane);
  v16bf aB = load_a_frag(arow + 32, lane);
  v16bf bB0 = load_b_frag(brow[0] + 32, lane);
  v16bf bB1 = load_b_frag(brow[1] + 32, lane);
  v16bf bB2 = load_b_frag(brow[2] + 32, lane);
  v16bf bB3 = load_b_frag(brow[3] + 32, lane);

#pragma unroll 2
  for (int kb = 0; kb < DD; kb += 64) {
    // tail reloads wrap to k=0 (masked, branch-free, always in-bounds; the
    // wrapped data is never consumed)
    const int kA = (kb + 64) & (DD - 1);
    const int kB = (kb + 96) & (DD - 1);

    acc[0] = wmma_bf16(aA, bA0, acc[0]);
    acc[1] = wmma_bf16(aA, bA1, acc[1]);
    acc[2] = wmma_bf16(aA, bA2, acc[2]);
    acc[3] = wmma_bf16(aA, bA3, acc[3]);
    aA = load_a_frag(arow + kA, lane);
    bA0 = load_b_frag(brow[0] + kA, lane);
    bA1 = load_b_frag(brow[1] + kA, lane);
    bA2 = load_b_frag(brow[2] + kA, lane);
    bA3 = load_b_frag(brow[3] + kA, lane);

    acc[0] = wmma_bf16(aB, bB0, acc[0]);
    acc[1] = wmma_bf16(aB, bB1, acc[1]);
    acc[2] = wmma_bf16(aB, bB2, acc[2]);
    acc[3] = wmma_bf16(aB, bB3, acc[3]);
    aB = load_a_frag(arow + kB, lane);
    bB0 = load_b_frag(brow[0] + kB, lane);
    bB1 = load_b_frag(brow[1] + kB, lane);
    bB2 = load_b_frag(brow[2] + kB, lane);
    bB3 = load_b_frag(brow[3] + kB, lane);
  }

  const int rbase = (lane & 16) ? 8 : 0;
#pragma unroll
  for (int j = 0; j < 4; ++j) {
    const int col = n0 + j * 16 + ln16;
    const float bv = bias[col];
    float* op = xw + (size_t)(m0 + rbase) * GG + col;
#pragma unroll
    for (int r = 0; r < 8; ++r) op[(size_t)r * GG] = acc[j][r] + bv;
  }
}

// ---------------- phase 2: persistent sequential LSTM scan ----------------
__global__ void __launch_bounds__(256, 1)
k_lstm_scan(const float* __restrict__ xw,          // [T*16][4096] f32 (bias folded)
            const unsigned short* __restrict__ ut, // [4096][1024] bf16 (U^T)
            const float* __restrict__ x,           // [16][T][D] f32
            unsigned short* __restrict__ hbf,      // 2 x [16][1024] bf16 (ping-pong)
            unsigned* __restrict__ bar,            // {count, epoch}
            float* __restrict__ out) {             // [16][T][D] f32
  __shared__ __align__(16) unsigned char smem[SMEM_BYTES];
  const int tid = threadIdx.x;
  const int lane = tid & 31;
  const int ln16 = lane & 15;
  const int wid = tid >> 5;
  const int c0 = (int)blockIdx.x * HCOLS;
  const int q = wid >> 1;    // gate handled by this wave (i,f,g,o)
  const int hh = wid & 1;    // 16-col half within the 32-col gate slice
  const int zcol = q * DD + c0 + hh * 16 + ln16;   // z column (N) of this lane
  const int brow = q * 32 + hh * 16 + ln16;        // LDS U-row for this lane

  const uint64_t utbase = (uint64_t)(uintptr_t)ut + (uint64_t)c0 * DD * 2ull;
  const unsigned lds0 = (unsigned)(uintptr_t)(&smem[0]);

  // register-resident cell state: 2 (row,col) slots per thread
  const int s0 = tid * 2;
  const int row0 = s0 >> 5, d0 = s0 & 31;
  const int row1 = (s0 + 1) >> 5, d1 = (s0 + 1) & 31;
  float cst0 = 0.0f, cst1 = 0.0f;

  // TDM prologue: chunk 0 -> buffer 0 (one DMA per WG, issued by wave 0)
  if (wid == 0) tdm_issue_u(utbase, lds0);

  for (int t = 0; t < TT; ++t) {
    const unsigned short* hread = hbf + (size_t)(t & 1) * (BB * DD);
    unsigned short* hwrite = hbf + (size_t)((t + 1) & 1) * (BB * DD);

    // acc = xW[t] tile (bias already folded in)
    v8f acc;
    {
      const float* xwp =
          xw + (size_t)(t * 16 + ((lane & 16) ? 8 : 0)) * GG + zcol;
#pragma unroll
      for (int r = 0; r < 8; ++r) acc[r] = xwp[(size_t)r * GG];
    }

    const unsigned short* hrow = hread + (size_t)ln16 * DD;

    // K loop: double-buffered TDM stream of the WG's U slice out of L2.
    for (int ci = 0; ci < NCHUNK; ++ci) {
      if (wid == 0) {
        const int cn = (ci + 1) & (NCHUNK - 1);  // wraps to chunk0 = prefetch for t+1
        tdm_issue_u(utbase + (uint64_t)(cn * KCHUNK) * 2ull,
                    lds0 + (unsigned)((cn & 1) * UBUF_BYTES));
        __builtin_amdgcn_s_wait_tensorcnt(1);    // current chunk landed
      }
      __syncthreads();
      const unsigned short* bb =
          (const unsigned short*)(smem + (ci & 1) * UBUF_BYTES + brow * ROWB);
      // issue all fragment loads (global A + LDS B) before the WMMA pair
      v16bf a0 = load_a_frag(hrow + ci * KCHUNK, lane);
      v16bf b0 = load_b_frag(bb, lane);
      v16bf a1 = load_a_frag(hrow + ci * KCHUNK + 32, lane);
      v16bf b1 = load_b_frag(bb + 32, lane);
      acc = wmma_bf16(a0, b0, acc);
      acc = wmma_bf16(a1, b1, acc);
      __syncthreads();  // buffer consumed before it is refilled
    }

    // z tile -> LDS (gate-major columns: q*32 + hh*16 + n)
    {
      float* zb = (float*)(smem + ZBUF_OFF);
      const int rbase = (lane & 16) ? 8 : 0;
      const int zc = wid * 16 + ln16;
#pragma unroll
      for (int r = 0; r < 8; ++r) zb[(rbase + r) * 128 + zc] = acc[r];
    }
    __syncthreads();

    // gates + state update + residual output (2 slots per thread)
    {
      const float* zb = (const float*)(smem + ZBUF_OFF);
      float iv0 = sigmoidf_(zb[row0 * 128 + d0]);
      float fv0 = sigmoidf_(zb[row0 * 128 + 32 + d0]);
      float gv0 = fmaxf(zb[row0 * 128 + 64 + d0], 0.0f);
      float ov0 = sigmoidf_(zb[row0 * 128 + 96 + d0]);
      cst0 = fv0 * cst0 + iv0 * gv0;
      float hv0 = ov0 * fmaxf(cst0, 0.0f);

      float iv1 = sigmoidf_(zb[row1 * 128 + d1]);
      float fv1 = sigmoidf_(zb[row1 * 128 + 32 + d1]);
      float gv1 = fmaxf(zb[row1 * 128 + 64 + d1], 0.0f);
      float ov1 = sigmoidf_(zb[row1 * 128 + 96 + d1]);
      cst1 = fv1 * cst1 + iv1 * gv1;
      float hv1 = ov1 * fmaxf(cst1, 0.0f);

      size_t oi0 = ((size_t)row0 * TT + t) * DD + (c0 + d0);
      size_t oi1 = ((size_t)row1 * TT + t) * DD + (c0 + d1);
      out[oi0] = hv0 + x[oi0];
      out[oi1] = hv1 + x[oi1];
      hwrite[row0 * DD + c0 + d0] = f2bf(hv0);
      hwrite[row1 * DD + c0 + d1] = f2bf(hv1);
    }

    // device-wide release/acquire epoch barrier before next step reads h
    __builtin_amdgcn_fence(__ATOMIC_RELEASE, "agent");  // global_wb
    __syncthreads();
    if (tid == 0) {
      const unsigned target = (unsigned)(t + 1);
      if (__hip_atomic_fetch_add(&bar[0], 1u, __ATOMIC_ACQ_REL,
                                 __HIP_MEMORY_SCOPE_AGENT) == NWG - 1) {
        __hip_atomic_store(&bar[0], 0u, __ATOMIC_RELAXED, __HIP_MEMORY_SCOPE_AGENT);
        __hip_atomic_store(&bar[1], target, __ATOMIC_RELEASE, __HIP_MEMORY_SCOPE_AGENT);
      } else {
        while (__hip_atomic_load(&bar[1], __ATOMIC_ACQUIRE,
                                 __HIP_MEMORY_SCOPE_AGENT) < target)
          __builtin_amdgcn_s_sleep(2);
      }
    }
    __builtin_amdgcn_s_cluster_barrier();  // NOP outside a cluster dispatch
    __syncthreads();
    __builtin_amdgcn_fence(__ATOMIC_ACQUIRE, "agent");  // global_inv
  }
}

// ---------------- host-side launch ----------------
extern "C" void kernel_launch(void* const* d_in, const int* in_sizes, int n_in,
                              void* d_out, int out_size, void* d_ws, size_t ws_size,
                              hipStream_t stream) {
  const float* x = (const float*)d_in[0];  // [16][1024][1024]
  const float* W = (const float*)d_in[1];  // [1024][4096]
  const float* U = (const float*)d_in[2];  // [1024][4096]
  const float* b = (const float*)d_in[3];  // [4096]
  float* out = (float*)d_out;              // [16][1024][1024]

  char* ws = (char*)d_ws;
  unsigned short* xb = (unsigned short*)(ws);                       // 32 MB bf16
  unsigned short* wt = (unsigned short*)(ws + (32ull << 20));       //  8 MB bf16
  unsigned short* ut = (unsigned short*)(ws + (40ull << 20));       //  8 MB bf16
  float* xw = (float*)(ws + (48ull << 20));                         // 256 MB f32
  unsigned short* hbf = (unsigned short*)(ws + (304ull << 20));     // 64 KB (x2 ping-pong)
  unsigned* bar = (unsigned*)(ws + (304ull << 20) + (128u << 10));  // barrier {count,epoch}

  k_init<<<64, 256, 0, stream>>>(hbf, bar);
  k_cvt_x<<<16384, 256, 0, stream>>>(x, xb);
  k_cvt_t<<<4096, 256, 0, stream>>>(W, wt);
  k_cvt_t<<<4096, 256, 0, stream>>>(U, ut);
  k_xw_gemm<<<8192, 256, 0, stream>>>(xb, wt, b, xw);
  k_lstm_scan<<<NWG, 256, 0, stream>>>(xw, ut, x, hbf, bar, out);
}
